// LinearAttention_38130719654507
// MI455X (gfx1250) — matrix operand order repaired and checked by reading, not compiled, not run on hardware
//
#include <hip/hip_runtime.h>
#include <hip/hip_bf16.h>
#include <math.h>

// Problem constants (from reference)
#define BQ     8
#define NQ     16384
#define DIMK   256            // input feature dim (K of GEMM)
#define DI     64             // DIM_INNER
#define NOUT   192            // 3 * DIM_INNER
#define HEADS  4
#define DH     16
#define ROWS_TOTAL (BQ * NQ)  // 131072

// GEMM tiling
#define TILE_M    128
#define KCHUNK    32
#define NKC       (DIMK / KCHUNK)   // 8
#define XS_STRIDE 40                // f16 per LDS x row (32 + 8 pad)
#define WT_LDS_STRIDE 264           // f16 per LDS Wt row (256 + 8 pad)
#define EV_STRIDE 68                // f32 per LDS exp(k)/v row (64 + 4 pad)

#define WT_BYTES  (NOUT * WT_LDS_STRIDE * 2)    // 101376
#define XS_BYTES  (2 * TILE_M * XS_STRIDE * 2)  // 20480 (double buffered)
// phase-2 arrays (exp(k), v : 2 * 128*68*4 = 69632 B) overlap wt/xs region
#define SMEM_BYTES (WT_BYTES + XS_BYTES + NOUT * 4 + TILE_M * 4)

#define CHUNKS_PER_B (NQ / TILE_M)  // 128 row-chunks per batch

typedef _Float16 v8h  __attribute__((ext_vector_type(8)));
typedef _Float16 v16h __attribute__((ext_vector_type(16)));
typedef float    v8f  __attribute__((ext_vector_type(8)));

static __device__ __forceinline__ v16h ld_frag16(const _Float16* p0, const _Float16* p1) {
    v8h lo = *(const v8h*)p0;
    v8h hi = *(const v8h*)p1;
    v16h r;
#pragma unroll
    for (int i = 0; i < 8; i++) { r[i] = lo[i]; r[i + 8] = hi[i]; }
    return r;
}

static __device__ __forceinline__ v8h pack8(float a, float b, float c, float d,
                                            float e, float f, float g, float h) {
    v8h r;
    r[0] = (_Float16)a; r[1] = (_Float16)b; r[2] = (_Float16)c; r[3] = (_Float16)d;
    r[4] = (_Float16)e; r[5] = (_Float16)f; r[6] = (_Float16)g; r[7] = (_Float16)h;
    return r;
}

// ---------------------------------------------------------------------------
// Kernel 0: one-time W (256x192 f32, row-major) -> Wt (192x256 f16, transposed)
// ---------------------------------------------------------------------------
__global__ __launch_bounds__(256) void wt_prep_kernel(
        const float* __restrict__ W, _Float16* __restrict__ wtg)
{
    int i = blockIdx.x * 256 + threadIdx.x;   // 0 .. 192*256-1
    int c = i >> 8;
    int k = i & 255;
    wtg[c * 256 + k] = (_Float16)W[k * NOUT + c];
}

// ---------------------------------------------------------------------------
// Kernel 1 (fused): qkv = x @ W + b via f16 WMMA (fp32 acc); then, without
// touching HBM, apply mask + exp to the k/v tiles held in accumulators and
// reduce this block's 128 rows into partial kv[h][d][e] / S[h][d].
// Only q columns are written to global.
// ---------------------------------------------------------------------------
__global__ __launch_bounds__(256) void qkv_gemm_kv_kernel(
        const float* __restrict__ x, const _Float16* __restrict__ wtg,
        const float* __restrict__ bias, const int* __restrict__ mask,
        float* __restrict__ qbuf, float* __restrict__ kvpart,
        float* __restrict__ spart)
{
    extern __shared__ __align__(16) char smem[];
    _Float16* wt     = (_Float16*)smem;                          // [192][264]
    _Float16* xs     = (_Float16*)(smem + WT_BYTES);             // [2][128][40]
    float*    bias_s = (float*)(smem + WT_BYTES + XS_BYTES);     // [192]
    int*      msk_s  = (int*)(smem + WT_BYTES + XS_BYTES + NOUT * 4); // [128]

    const int  t    = threadIdx.x;
    const int  lane = t & 31;
    const int  wave = t >> 5;
    const int  half = lane >> 4;
    const int  l16  = lane & 15;
    const long rowBase = (long)blockIdx.x * TILE_M;   // 128 rows, one batch b

    // --- one-time: Wt f16 -> LDS via b128 copies (24 per thread) ---
#pragma unroll
    for (int i = 0; i < 24; i++) {
        int idx = t + i * 256;
        int c   = idx >> 5;        // 32 x 16B chunks per 512B row
        int ch  = idx & 31;
        *(v8h*)(wt + c * WT_LDS_STRIDE + ch * 8) = *(const v8h*)(wtg + c * 256 + ch * 8);
    }
    if (t < NOUT)   bias_s[t] = bias[t];
    if (t < TILE_M) msk_s[t]  = mask[rowBase + t];

    // per-thread x staging geometry: thread owns half a row (16 floats)
    const int    sr   = t >> 1;
    const int    sc0  = (t & 1) * 16;
    const float* srow = x + (rowBase + sr) * DIMK + sc0;
    _Float16*    dst0 = xs + sr * XS_STRIDE + sc0;

    // preload chunk 0
    {
        const float4* s4 = (const float4*)srow;
        float4 f0 = s4[0], f1 = s4[1], f2 = s4[2], f3 = s4[3];
        *(v8h*)dst0       = pack8(f0.x, f0.y, f0.z, f0.w, f1.x, f1.y, f1.z, f1.w);
        *((v8h*)dst0 + 1) = pack8(f2.x, f2.y, f2.z, f2.w, f3.x, f3.y, f3.z, f3.w);
    }
    __syncthreads();

    v8f acc[12];
#pragma unroll
    for (int i = 0; i < 12; i++)
#pragma unroll
        for (int r = 0; r < 8; r++) acc[i][r] = 0.0f;

#pragma unroll
    for (int kc = 0; kc < NKC; kc++) {
        const _Float16* xbuf = xs + (kc & 1) * (TILE_M * XS_STRIDE);

        // issue next chunk's global loads early (latency hidden behind WMMAs)
        float4 f0, f1, f2, f3;
        if (kc + 1 < NKC) {
            const float4* s4 = (const float4*)(srow + (kc + 1) * KCHUNK);
            f0 = s4[0]; f1 = s4[1]; f2 = s4[2]; f3 = s4[3];
            if (kc + 2 < NKC)
                __builtin_prefetch(srow + (kc + 2) * KCHUNK, 0, 1);
        }

        const _Float16* arow = xbuf + (wave * 16 + l16) * XS_STRIDE;
        v16h a = ld_frag16(arow + half * 8, arow + 16 + half * 8);
        const int kof = kc * KCHUNK + half * 8;
#pragma unroll
        for (int ct = 0; ct < 12; ct++) {
            const _Float16* brow = wt + (ct * 16 + l16) * WT_LDS_STRIDE + kof;
            v16h b = ld_frag16(brow, brow + 16);
            acc[ct] = __builtin_amdgcn_wmma_f32_16x16x32_f16(
                false, a, false, b, (short)0, acc[ct], false, false);
        }

        // store next chunk into the other buffer, then one barrier
        if (kc + 1 < NKC) {
            _Float16* d = xs + ((kc + 1) & 1) * (TILE_M * XS_STRIDE) + sr * XS_STRIDE + sc0;
            *(v8h*)d       = pack8(f0.x, f0.y, f0.z, f0.w, f1.x, f1.y, f1.z, f1.w);
            *((v8h*)d + 1) = pack8(f2.x, f2.y, f2.z, f2.w, f3.x, f3.y, f3.z, f3.w);
            __syncthreads();
        }
    }

    // --- q epilogue: add bias, write only q columns (ct 0..3) ---
#pragma unroll
    for (int ct = 0; ct < 4; ct++) {
        int   col = ct * 16 + l16;
        float bb  = bias_s[col];
        long  rg  = rowBase + wave * 16 + half * 8;
#pragma unroll
        for (int r = 0; r < 8; r++)
            qbuf[(rg + r) * DI + col] = acc[ct][r] + bb;
    }

    // --- fused kv reduction: wt/xs LDS is dead now, reuse it ---
    __syncthreads();   // all WMMA-feeding ds reads complete before overwrite
    float* eks = (float*)smem;                   // [128][68]  exp(k) (masked)
    float* vvs = eks + TILE_M * EV_STRIDE;       // [128][68]  v (masked)

#pragma unroll
    for (int r = 0; r < 8; r++) {
        int  row = wave * 16 + half * 8 + r;
        bool m   = msk_s[row] != 0;
#pragma unroll
        for (int h = 0; h < HEADS; h++) {
            float kval = acc[4 + h][r] + bias_s[64  + h * DH + l16];
            float vval = acc[8 + h][r] + bias_s[128 + h * DH + l16];
            eks[row * EV_STRIDE + h * DH + l16] = m ? __expf(kval) : 0.0f;
            vvs[row * EV_STRIDE + h * DH + l16] = m ? vval : 0.0f;
        }
    }
    __syncthreads();

    // thread owns (head, d, e-quad): 1 b32 + 1 b128 DS read per row, 4 FMAs
    const int h2 = t >> 6;
    const int dd = (t >> 2) & 15;
    const int e0 = (t & 3) * 4;
    float4 kvq = make_float4(0.f, 0.f, 0.f, 0.f);
    float  ss  = 0.0f;
#pragma unroll 4
    for (int i = 0; i < TILE_M; i++) {
        float  e  = eks[i * EV_STRIDE + h2 * DH + dd];
        float4 vv = *(const float4*)&vvs[i * EV_STRIDE + h2 * DH + e0];
        kvq.x = fmaf(e, vv.x, kvq.x);
        kvq.y = fmaf(e, vv.y, kvq.y);
        kvq.z = fmaf(e, vv.z, kvq.z);
        kvq.w = fmaf(e, vv.w, kvq.w);
        ss += e;
    }
    *(float4*)&kvpart[(long)blockIdx.x * (HEADS * 256) + h2 * 256 + dd * 16 + e0] = kvq;
    if ((t & 3) == 0) spart[blockIdx.x * (HEADS * DH) + h2 * DH + dd] = ss;
}

// ---------------------------------------------------------------------------
// Kernel 2: fold 128 row-chunk partials per (b,h), normalize by S[d]
// ---------------------------------------------------------------------------
__global__ __launch_bounds__(256) void kv_combine_kernel(
        const float* __restrict__ kvpart, const float* __restrict__ spart,
        float* __restrict__ kvn)
{
    const int bh  = blockIdx.x;   // 0..31
    const int b   = bh >> 2;
    const int h   = bh & 3;
    const int t   = threadIdx.x;  // (d,e) = (t>>4, t&15)
    const int dmy = t >> 4;
    float kv = 0.0f, s = 0.0f;
#pragma unroll 8
    for (int c = 0; c < CHUNKS_PER_B; c++) {
        long blk = (long)(b * CHUNKS_PER_B + c);
        kv += kvpart[blk * (HEADS * 256) + h * 256 + t];
        s  += spart [blk * (HEADS * DH)  + h * DH  + dmy];
    }
    kvn[bh * 256 + t] = (s > 0.0f) ? (kv / s) : 0.0f;
}

// ---------------------------------------------------------------------------
// Kernel 3: out[n, h*16+e] = sum_d softmax_row(q)[d] * kvn[b,h][d][e]
// One thread per (row, head); kvn cached in LDS; float4 q reads / out writes.
// ---------------------------------------------------------------------------
__global__ __launch_bounds__(256) void out_kernel(
        const float* __restrict__ qbuf, const float* __restrict__ kvn,
        float* __restrict__ out)
{
    __shared__ float kv_s[HEADS * 256];
    const int  t       = threadIdx.x;
    const long rowBase = (long)blockIdx.x * 64;     // 64 rows per block, one b
    const int  b       = (int)(rowBase / NQ);
#pragma unroll
    for (int i = 0; i < 4; i++)
        kv_s[t + i * 256] = kvn[b * (HEADS * 256) + t + i * 256];
    __syncthreads();

    const int  lr  = t >> 2;
    const int  h   = t & 3;
    const long row = rowBase + lr;

    float qv[16];
    const float4* q4 = (const float4*)(qbuf + row * DI + h * DH);
#pragma unroll
    for (int j = 0; j < 4; j++) ((float4*)qv)[j] = q4[j];

    float mx = -3.0e38f;
#pragma unroll
    for (int d = 0; d < 16; d++) mx = fmaxf(mx, qv[d]);
    float s = 0.0f;
#pragma unroll
    for (int d = 0; d < 16; d++) { qv[d] = __expf(qv[d] - mx); s += qv[d]; }
    const float inv = 1.0f / s;

    const float* kvh = &kv_s[h * 256];
    float o[16];
#pragma unroll
    for (int e = 0; e < 16; e++) o[e] = 0.0f;
#pragma unroll
    for (int d = 0; d < 16; d++) {
        float qd = qv[d] * inv;
#pragma unroll
        for (int e = 0; e < 16; e++) o[e] = fmaf(qd, kvh[d * 16 + e], o[e]);
    }
    float4* o4 = (float4*)(out + row * DI + h * DH);
#pragma unroll
    for (int j = 0; j < 4; j++) o4[j] = ((float4*)o)[j];
}

extern "C" void kernel_launch(void* const* d_in, const int* in_sizes, int n_in,
                              void* d_out, int out_size, void* d_ws, size_t ws_size,
                              hipStream_t stream)
{
    const float* x    = (const float*)d_in[0];  // (B, N, 256) f32
    const int*   mask = (const int*)  d_in[1];  // (B, N) bool -> int per harness
    const float* W    = (const float*)d_in[2];  // (256, 192) f32
    const float* bias = (const float*)d_in[3];  // (192,) f32
    float*       out  = (float*)d_out;          // (B, N, 64) f32

    const int nblocks = ROWS_TOTAL / TILE_M;    // 1024

    // Workspace layout (floats): q | kvpart | spart | kvn | Wt(f16)
    float* ws     = (float*)d_ws;
    float* qbuf   = ws;                                          // 8,388,608 f
    float* kvpart = qbuf   + (size_t)ROWS_TOTAL * DI;            // 1,048,576 f
    float* spart  = kvpart + (size_t)nblocks * HEADS * 256;      //    65,536 f
    float* kvn    = spart  + (size_t)nblocks * HEADS * DH;       //     8,192 f
    _Float16* wtg = (_Float16*)(kvn + (size_t)BQ * HEADS * 256); //    49,152 h

    wt_prep_kernel    <<<NOUT,             256, 0,          stream>>>(W, wtg);
    qkv_gemm_kv_kernel<<<nblocks,          256, SMEM_BYTES, stream>>>(
        x, wtg, bias, mask, qbuf, kvpart, spart);
    kv_combine_kernel <<<BQ * HEADS,       256, 0,          stream>>>(kvpart, spart, kvn);
    out_kernel        <<<ROWS_TOTAL / 64,  256, 0,          stream>>>(qbuf, kvn, out);
}
